// MambaAttention_29463475651339
// MI455X (gfx1250) — matrix operand order repaired and checked by reading, not compiled, not run on hardware
//
#include <hip/hip_runtime.h>
#include <hip/hip_bf16.h>
#include <math.h>

// ---------------- problem constants ----------------
#define Bz    4
#define Nn    4096
#define Dm    1024
#define Ss    16
#define Rr    64
#define XP    96          // R + 2S
#define MTOK  (Bz*Nn)     // 16384 rows
#define NCH   32          // scan chunks
#define CL    128         // chunk length  (NCH*CL == Nn)
#define SA    40          // LDS row stride (bf16 elems): 64B data + 16B pad

typedef __attribute__((ext_vector_type(16))) __bf16 v16bf;
typedef __attribute__((ext_vector_type(8)))  float  v8f;
typedef __attribute__((ext_vector_type(4)))  unsigned int u32x4;
typedef __attribute__((ext_vector_type(8)))  int          i32x8;
typedef __attribute__((ext_vector_type(4)))  int          i32x4;

#if defined(__has_builtin)
#if __has_builtin(__builtin_amdgcn_tensor_load_to_lds) && __has_builtin(__builtin_amdgcn_s_wait_tensorcnt)
#define USE_TDM 1
#endif
#endif
#ifndef USE_TDM
#define USE_TDM 0
#endif

union FragAB { v16bf v; unsigned u[8]; };

// round-to-nearest-even f32->bf16 (used for persistent activations)
__device__ __forceinline__ unsigned short f2bf(float f) {
  union { float f; unsigned u; } c; c.f = f;
  unsigned u = c.u;
  return (unsigned short)((u + 0x7FFFu + ((u >> 16) & 1u)) >> 16);
}
// truncating pack of two f32 -> bf16x2 (cheap; LDS staging only)
__device__ __forceinline__ unsigned pack_bf2_rz(float a, float b) {
  union { float f; unsigned u; } ca, cb; ca.f = a; cb.f = b;
  return (ca.u >> 16) | (cb.u & 0xFFFF0000u);
}
__device__ __forceinline__ unsigned short f2bf_rz(float a) {
  union { float f; unsigned u; } c; c.f = a;
  return (unsigned short)(c.u >> 16);
}

#if USE_TDM
// TDM: DMA a (tile_rows x 32 bf16) tile, row stride row_stride_elems (bf16),
// into LDS at lds_addr with 16B padding after each 64B row (-> SA=40 layout).
__device__ __forceinline__ void tdm_load_tile_bf16(
    unsigned lds_addr, const void* gptr,
    unsigned tile_rows, unsigned row_stride_elems)
{
  unsigned long long ga = (unsigned long long)(size_t)gptr;
  u32x4 g0;
  g0[0] = 1u;                                   // count=1, user descriptor
  g0[1] = lds_addr;                             // LDS byte address
  g0[2] = (unsigned)ga;                         // global_addr[31:0]
  g0[3] = (unsigned)((ga >> 32) & 0x1FFFFFFu)   // global_addr[56:32]
        | (2u << 30);                           // type=2 ("image")
  i32x8 g1;
  // word0: wg_mask=0 | data_size=1(2B)<<16 | pad_enable<<20
  //        | pad_interval=3(16 DW)<<22 | pad_amount=3(4 DW)<<25
  g1[0] = (1 << 16) | (1 << 20) | (3 << 22) | (3 << 25);
  const unsigned td = 1u << 20;                 // generous tensor dims (in-bounds tiles)
  g1[1] = (int)((td & 0xFFFFu) << 16);          // [47:32]=ab_addr=0, [63:48]=td0.lo
  g1[2] = (int)((td >> 16) & 0xFFFFu)           // [79:64]=td0.hi
        | (int)((td & 0xFFFFu) << 16);          // [95:80]=td1.lo
  g1[3] = (int)((td >> 16) & 0xFFFFu)           // [111:96]=td1.hi
        | (int)(32u << 16);                     // [127:112]=tile_dim0=32 elems
  g1[4] = (int)tile_rows;                       // [143:128]=tile_dim1; tile_dim2=0
  g1[5] = (int)row_stride_elems;                // tensor_dim0_stride[31:0]
  g1[6] = 0;                                    // stride hi | dim1_stride.lo
  g1[7] = 0;
  i32x4 z4 = {0, 0, 0, 0};
  i32x8 z8 = {0, 0, 0, 0, 0, 0, 0, 0};
  __builtin_amdgcn_tensor_load_to_lds(g0, g1, z4, z4, z8, 0);
}
#endif

// =====================================================================
// GEMM1: xz = values @ in_proj_w + b   (16384x1024)x(1024x2048)
// epilogue: cols<1024 -> x_ssm (f32 + bf16 copy); cols>=1024 -> silu(z)
// =====================================================================
__global__ __launch_bounds__(256) void gemm1_inproj(
    const float* __restrict__ V, const float* __restrict__ Wp,
    const float* __restrict__ bias,
    float* __restrict__ xssm_f, unsigned short* __restrict__ xssm_h,
    float* __restrict__ siluz)
{
  __shared__ unsigned short As[128 * SA];
  __shared__ unsigned short BsT[128 * SA];   // transposed: [n][k]
  const int tid  = threadIdx.x;
  const int lane = tid & 31, half = lane >> 4, lm = lane & 15;
  const int wid  = tid >> 5, wm = wid & 3, wn = wid >> 2;   // 4x2 waves
  const int m0 = blockIdx.y * 128, n0 = blockIdx.x * 128;

  v8f zero = {0.f,0.f,0.f,0.f,0.f,0.f,0.f,0.f};
  v8f acc[2][4];
  for (int mi = 0; mi < 2; ++mi)
    for (int ni = 0; ni < 4; ++ni) acc[mi][ni] = zero;

  for (int k0 = 0; k0 < Dm; k0 += 32) {
    // stage A tile (128x32 f32 -> bf16, truncating pack)
    #pragma unroll
    for (int it = 0; it < 4; ++it) {
      int li = it * 256 + tid;
      int r = li >> 3, c4 = (li & 7) << 2;
      float4 a4 = *(const float4*)(V + (size_t)(m0 + r) * Dm + k0 + c4);
      uint2 p; p.x = pack_bf2_rz(a4.x, a4.y); p.y = pack_bf2_rz(a4.z, a4.w);
      *(uint2*)&As[r * SA + c4] = p;
    }
    // stage B tile transposed (32x128 f32 -> BsT[n][k] bf16)
    #pragma unroll
    for (int it = 0; it < 4; ++it) {
      int li = it * 256 + tid;
      int kk = li >> 5, n4 = (li & 31) << 2;
      float4 b4 = *(const float4*)(Wp + (size_t)(k0 + kk) * (2 * Dm) + n0 + n4);
      BsT[(n4 + 0) * SA + kk] = f2bf_rz(b4.x);
      BsT[(n4 + 1) * SA + kk] = f2bf_rz(b4.y);
      BsT[(n4 + 2) * SA + kk] = f2bf_rz(b4.z);
      BsT[(n4 + 3) * SA + kk] = f2bf_rz(b4.w);
    }
    if (k0 + 32 < Dm)   // pull next A tile toward the caches
      __builtin_prefetch(V + (size_t)(m0 + (tid >> 1)) * Dm + k0 + 32, 0, 0);
    __syncthreads();

    FragAB a[2];
    #pragma unroll
    for (int mi = 0; mi < 2; ++mi) {
      const unsigned short* ap = &As[(wm * 32 + mi * 16 + lm) * SA + 16 * half];
      #pragma unroll
      for (int i = 0; i < 4; ++i) a[mi].u[i]     = *(const unsigned*)(ap + 2 * i);
      #pragma unroll
      for (int i = 0; i < 4; ++i) a[mi].u[4 + i] = *(const unsigned*)(ap + 8 + 2 * i);
    }
    #pragma unroll
    for (int ni = 0; ni < 4; ++ni) {
      FragAB bb;
      const unsigned short* bp = &BsT[(wn * 64 + ni * 16 + lm) * SA + 16 * half];
      #pragma unroll
      for (int i = 0; i < 8; ++i) bb.u[i] = *(const unsigned*)(bp + 2 * i);
      #pragma unroll
      for (int mi = 0; mi < 2; ++mi)
        acc[mi][ni] = __builtin_amdgcn_wmma_f32_16x16x32_bf16(
            false, a[mi].v, false, bb.v, (short)0, acc[mi][ni], false, false);
    }
    __syncthreads();
  }

  #pragma unroll
  for (int mi = 0; mi < 2; ++mi)
    #pragma unroll
    for (int ni = 0; ni < 4; ++ni)
      #pragma unroll
      for (int i = 0; i < 8; ++i) {
        int grow = m0 + wm * 32 + mi * 16 + half * 8 + i;
        int gcol = n0 + wn * 64 + ni * 16 + lm;
        float v = acc[mi][ni][i] + bias[gcol];
        if (gcol < Dm) {
          xssm_f[(size_t)grow * Dm + gcol] = v;
          xssm_h[(size_t)grow * Dm + gcol] = f2bf(v);
        } else {
          siluz[(size_t)grow * Dm + (gcol - Dm)] = v / (1.0f + __expf(-v));
        }
      }
}

// =====================================================================
// GEMM2: x_dbl = x_ssm @ x_proj_w + b   (16384x1024)x(1024x96)
// A tile staged by the Tensor Data Mover (bf16, straight strided copy).
// =====================================================================
__global__ __launch_bounds__(256) void gemm2_xproj(
    const unsigned short* __restrict__ Xh, const float* __restrict__ Wp,
    const float* __restrict__ bias,
    float* __restrict__ xdbl, unsigned short* __restrict__ dtlow_h)
{
  __shared__ unsigned short As[128 * SA];
  __shared__ unsigned short BsT[XP * SA];
  const int tid  = threadIdx.x;
  const int lane = tid & 31, half = lane >> 4, lm = lane & 15;
  const int wid  = tid >> 5, wm = wid & 3, wn = wid >> 2;   // 4x2 waves, WN=48
  const int m0 = blockIdx.y * 128;
#if USE_TDM
  const unsigned asAddr = (unsigned)(size_t)(void*)As;
#endif

  v8f zero = {0.f,0.f,0.f,0.f,0.f,0.f,0.f,0.f};
  v8f acc[2][3];
  for (int mi = 0; mi < 2; ++mi)
    for (int ni = 0; ni < 3; ++ni) acc[mi][ni] = zero;

  for (int k0 = 0; k0 < Dm; k0 += 32) {
#if USE_TDM
    if (tid < 32)     // one wave issues the tensor DMA for the whole A tile
      tdm_load_tile_bf16(asAddr, Xh + (size_t)m0 * Dm + k0, 128, Dm);
#else
    #pragma unroll
    for (int it = 0; it < 2; ++it) {
      int li = it * 256 + tid;
      int r = li >> 2, c8 = (li & 3) << 3;
      *(uint4*)&As[r * SA + c8] =
          *(const uint4*)(Xh + (size_t)(m0 + r) * Dm + k0 + c8);
    }
#endif
    #pragma unroll
    for (int it = 0; it < 3; ++it) {        // B tile 32x96, transpose+convert
      int li = it * 256 + tid;
      int kk = li / 24, n4 = (li % 24) * 4;
      float4 b4 = *(const float4*)(Wp + (size_t)(k0 + kk) * XP + n4);
      BsT[(n4 + 0) * SA + kk] = f2bf_rz(b4.x);
      BsT[(n4 + 1) * SA + kk] = f2bf_rz(b4.y);
      BsT[(n4 + 2) * SA + kk] = f2bf_rz(b4.z);
      BsT[(n4 + 3) * SA + kk] = f2bf_rz(b4.w);
    }
#if USE_TDM
    if (tid < 32) __builtin_amdgcn_s_wait_tensorcnt(0);
#endif
    __syncthreads();

    FragAB a[2];
    #pragma unroll
    for (int mi = 0; mi < 2; ++mi) {
      const unsigned short* ap = &As[(wm * 32 + mi * 16 + lm) * SA + 16 * half];
      #pragma unroll
      for (int i = 0; i < 4; ++i) a[mi].u[i]     = *(const unsigned*)(ap + 2 * i);
      #pragma unroll
      for (int i = 0; i < 4; ++i) a[mi].u[4 + i] = *(const unsigned*)(ap + 8 + 2 * i);
    }
    #pragma unroll
    for (int ni = 0; ni < 3; ++ni) {
      FragAB bb;
      const unsigned short* bp = &BsT[(wn * 48 + ni * 16 + lm) * SA + 16 * half];
      #pragma unroll
      for (int i = 0; i < 8; ++i) bb.u[i] = *(const unsigned*)(bp + 2 * i);
      #pragma unroll
      for (int mi = 0; mi < 2; ++mi)
        acc[mi][ni] = __builtin_amdgcn_wmma_f32_16x16x32_bf16(
            false, a[mi].v, false, bb.v, (short)0, acc[mi][ni], false, false);
    }
    __syncthreads();
  }

  #pragma unroll
  for (int mi = 0; mi < 2; ++mi)
    #pragma unroll
    for (int ni = 0; ni < 3; ++ni)
      #pragma unroll
      for (int i = 0; i < 8; ++i) {
        int grow = m0 + wm * 32 + mi * 16 + half * 8 + i;
        int gcol = wn * 48 + ni * 16 + lm;
        float v = acc[mi][ni][i] + bias[gcol];
        xdbl[(size_t)grow * XP + gcol] = v;
        if (gcol < Rr) dtlow_h[(size_t)grow * Rr + gcol] = f2bf(v);
      }
}

// =====================================================================
// GEMM3: dt = softplus(dt_low @ dt_proj_w + b)  (16384x64)x(64x1024)
// A tile staged by TDM as well (bf16 source, row stride 64).
// =====================================================================
__global__ __launch_bounds__(256) void gemm3_dtproj(
    const unsigned short* __restrict__ Lh, const float* __restrict__ Wp,
    const float* __restrict__ bias, float* __restrict__ dtb)
{
  __shared__ unsigned short As[128 * SA];
  __shared__ unsigned short BsT[128 * SA];
  const int tid  = threadIdx.x;
  const int lane = tid & 31, half = lane >> 4, lm = lane & 15;
  const int wid  = tid >> 5, wm = wid & 3, wn = wid >> 2;
  const int m0 = blockIdx.y * 128, n0 = blockIdx.x * 128;
#if USE_TDM
  const unsigned asAddr = (unsigned)(size_t)(void*)As;
#endif

  v8f zero = {0.f,0.f,0.f,0.f,0.f,0.f,0.f,0.f};
  v8f acc[2][4];
  for (int mi = 0; mi < 2; ++mi)
    for (int ni = 0; ni < 4; ++ni) acc[mi][ni] = zero;

  for (int k0 = 0; k0 < Rr; k0 += 32) {
#if USE_TDM
    if (tid < 32)
      tdm_load_tile_bf16(asAddr, Lh + (size_t)m0 * Rr + k0, 128, Rr);
#else
    #pragma unroll
    for (int it = 0; it < 2; ++it) {
      int li = it * 256 + tid;
      int r = li >> 2, c8 = (li & 3) << 3;
      *(uint4*)&As[r * SA + c8] =
          *(const uint4*)(Lh + (size_t)(m0 + r) * Rr + k0 + c8);
    }
#endif
    #pragma unroll
    for (int it = 0; it < 4; ++it) {        // B tile 32x128 f32 -> bf16 T
      int li = it * 256 + tid;
      int kk = li >> 5, n4 = (li & 31) << 2;
      float4 b4 = *(const float4*)(Wp + (size_t)(k0 + kk) * Dm + n0 + n4);
      BsT[(n4 + 0) * SA + kk] = f2bf_rz(b4.x);
      BsT[(n4 + 1) * SA + kk] = f2bf_rz(b4.y);
      BsT[(n4 + 2) * SA + kk] = f2bf_rz(b4.z);
      BsT[(n4 + 3) * SA + kk] = f2bf_rz(b4.w);
    }
#if USE_TDM
    if (tid < 32) __builtin_amdgcn_s_wait_tensorcnt(0);
#endif
    __syncthreads();

    FragAB a[2];
    #pragma unroll
    for (int mi = 0; mi < 2; ++mi) {
      const unsigned short* ap = &As[(wm * 32 + mi * 16 + lm) * SA + 16 * half];
      #pragma unroll
      for (int i = 0; i < 4; ++i) a[mi].u[i]     = *(const unsigned*)(ap + 2 * i);
      #pragma unroll
      for (int i = 0; i < 4; ++i) a[mi].u[4 + i] = *(const unsigned*)(ap + 8 + 2 * i);
    }
    #pragma unroll
    for (int ni = 0; ni < 4; ++ni) {
      FragAB bb;
      const unsigned short* bp = &BsT[(wn * 64 + ni * 16 + lm) * SA + 16 * half];
      #pragma unroll
      for (int i = 0; i < 8; ++i) bb.u[i] = *(const unsigned*)(bp + 2 * i);
      #pragma unroll
      for (int mi = 0; mi < 2; ++mi)
        acc[mi][ni] = __builtin_amdgcn_wmma_f32_16x16x32_bf16(
            false, a[mi].v, false, bb.v, (short)0, acc[mi][ni], false, false);
    }
    __syncthreads();
  }

  #pragma unroll
  for (int mi = 0; mi < 2; ++mi)
    #pragma unroll
    for (int ni = 0; ni < 4; ++ni)
      #pragma unroll
      for (int i = 0; i < 8; ++i) {
        int grow = m0 + wm * 32 + mi * 16 + half * 8 + i;
        int gcol = n0 + wn * 64 + ni * 16 + lm;
        float v = acc[mi][ni][i] + bias[gcol];
        v = (v > 20.0f) ? v : log1pf(__expf(v));     // softplus
        dtb[(size_t)grow * Dm + gcol] = v;
      }
}

// =====================================================================
// Scan pass 1: per (b, chunk, d) compute local Y, W[s], P[s], h_end[s]
// =====================================================================
__global__ __launch_bounds__(256) void scan_part1(
    const float* __restrict__ dtb, const float* __restrict__ xssm,
    const float* __restrict__ siluz, const float* __restrict__ xdbl,
    const float* __restrict__ A_log, const float* __restrict__ Dp,
    float* __restrict__ Yb, float* __restrict__ Wb,
    float* __restrict__ Pb, float* __restrict__ Hb)
{
  __shared__ float BC[CL][32];                 // B_t[0..15] | C_t[0..15]
  const int tid = threadIdx.x;
  const int b = blockIdx.z, ch = blockIdx.y, d = blockIdx.x * 256 + tid;

  #pragma unroll
  for (int it = 0; it < 16; ++it) {
    int li = it * 256 + tid;
    int t = li >> 5, s = li & 31;
    BC[t][s] = xdbl[(size_t)(b * Nn + ch * CL + t) * XP + Rr + s];
  }
  __syncthreads();

  float Ar[Ss], h[Ss], cum[Ss], W[Ss];
  #pragma unroll
  for (int s = 0; s < Ss; ++s) {
    Ar[s]  = -__expf(A_log[d * Ss + s]);
    h[s]   = 0.0f; cum[s] = 1.0f; W[s] = 0.0f;
  }
  const float Dd = Dp[d];
  float Y = 0.0f;

  for (int tt = 0; tt < CL; ++tt) {
    size_t idx = (size_t)(b * Nn + ch * CL + tt) * Dm + d;
    float dtv = dtb[idx], xv = xssm[idx], szv = siluz[idx];
    float dx = dtv * xv;
    float yl = Dd * xv;
    #pragma unroll
    for (int s = 0; s < Ss; ++s) {
      float dA = __expf(Ar[s] * dtv);
      cum[s] *= dA;
      h[s] = fmaf(dA, h[s], dx * BC[tt][s]);
      float Ct = BC[tt][16 + s];
      yl = fmaf(Ct, h[s], yl);
      W[s] = fmaf(szv * Ct, cum[s], W[s]);
    }
    Y = fmaf(szv, yl, Y);
  }

  Yb[(size_t)(b * NCH + ch) * Dm + d] = Y;
  #pragma unroll
  for (int s = 0; s < Ss; ++s) {
    size_t o = (size_t)((b * NCH + ch) * Ss + s) * Dm + d;
    Wb[o] = W[s]; Pb[o] = cum[s]; Hb[o] = h[s];
  }
}

// Scan pass 2: serial combine over 32 chunks per (b,d)
__global__ void scan_part2(
    const float* __restrict__ Yb, const float* __restrict__ Wb,
    const float* __restrict__ Pb, const float* __restrict__ Hb,
    float* __restrict__ ctx)
{
  int gid = blockIdx.x * 256 + threadIdx.x;    // 0..4095
  int b = gid >> 10, d = gid & 1023;
  float h[Ss];
  #pragma unroll
  for (int s = 0; s < Ss; ++s) h[s] = 0.0f;
  float acc = 0.0f;
  for (int ch = 0; ch < NCH; ++ch) {
    acc += Yb[(size_t)(b * NCH + ch) * Dm + d];
    #pragma unroll
    for (int s = 0; s < Ss; ++s) {
      size_t o = (size_t)((b * NCH + ch) * Ss + s) * Dm + d;
      acc = fmaf(Wb[o], h[s], acc);
      h[s] = fmaf(Pb[o], h[s], Hb[o]);
    }
  }
  ctx[b * Dm + d] = acc * (1.0f / (float)Nn);
}

// LayerNorm over D per batch row
__global__ __launch_bounds__(256) void ln_kernel(
    const float* __restrict__ ctx, const float* __restrict__ gamma,
    const float* __restrict__ beta, float* __restrict__ normed)
{
  __shared__ float red[256];
  int b = blockIdx.x, tid = threadIdx.x;
  float lv[4]; float s = 0.0f;
  #pragma unroll
  for (int i = 0; i < 4; ++i) { lv[i] = ctx[b * Dm + tid + i * 256]; s += lv[i]; }
  red[tid] = s; __syncthreads();
  for (int off = 128; off > 0; off >>= 1) {
    if (tid < off) red[tid] += red[tid + off];
    __syncthreads();
  }
  float mu = red[0] * (1.0f / Dm); __syncthreads();
  float sq = 0.0f;
  #pragma unroll
  for (int i = 0; i < 4; ++i) { float dd = lv[i] - mu; sq += dd * dd; }
  red[tid] = sq; __syncthreads();
  for (int off = 128; off > 0; off >>= 1) {
    if (tid < off) red[tid] += red[tid + off];
    __syncthreads();
  }
  float inv = rsqrtf(red[0] * (1.0f / Dm) + 1e-5f);
  #pragma unroll
  for (int i = 0; i < 4; ++i) {
    int d = tid + i * 256;
    normed[b * Dm + d] = (lv[i] - mu) * inv * gamma[d] + beta[d];
  }
}

// Tiny out-projection: (B x D) @ (D x D)
__global__ __launch_bounds__(256) void outproj_kernel(
    const float* __restrict__ normed, const float* __restrict__ Wp,
    const float* __restrict__ bias, float* __restrict__ outrow)
{
  __shared__ float nr[Dm];
  int b = blockIdx.x >> 2, jb = blockIdx.x & 3, tid = threadIdx.x;
  #pragma unroll
  for (int i = 0; i < 4; ++i) nr[tid + i * 256] = normed[b * Dm + tid + i * 256];
  __syncthreads();
  int j = jb * 256 + tid;
  float acc = bias[j];
  for (int i = 0; i < Dm; ++i) acc = fmaf(nr[i], Wp[(size_t)i * Dm + j], acc);
  outrow[b * Dm + j] = acc;
}

// Broadcast (B,D) row to (B,N,D) with float4 stores (the big write stream)
__global__ __launch_bounds__(256) void bcast_kernel(
    const float4* __restrict__ rowv, float4* __restrict__ out)
{
  const long long perB  = (long long)Nn * (Dm / 4);
  const long long total = (long long)Bz * perB;
  for (long long idx = (long long)blockIdx.x * 256 + threadIdx.x;
       idx < total; idx += (long long)gridDim.x * 256) {
    int b  = (int)(idx / perB);
    int d4 = (int)(idx & 255);                 // Dm/4 == 256, perB % 256 == 0
    out[idx] = rowv[b * 256 + d4];
  }
}

// =====================================================================
extern "C" void kernel_launch(void* const* d_in, const int* in_sizes, int n_in,
                              void* d_out, int out_size, void* d_ws, size_t ws_size,
                              hipStream_t stream) {
  const float* values     = (const float*)d_in[2];
  const float* in_proj_w  = (const float*)d_in[3];
  const float* in_proj_b  = (const float*)d_in[4];
  const float* x_proj_w   = (const float*)d_in[5];
  const float* x_proj_b   = (const float*)d_in[6];
  const float* dt_proj_w  = (const float*)d_in[7];
  const float* dt_proj_b  = (const float*)d_in[8];
  const float* A_log      = (const float*)d_in[9];
  const float* D_param    = (const float*)d_in[10];
  const float* out_proj_w = (const float*)d_in[11];
  const float* out_proj_b = (const float*)d_in[12];
  const float* ln_gamma   = (const float*)d_in[13];
  const float* ln_beta    = (const float*)d_in[14];
  float* out = (float*)d_out;

  char* wsb = (char*)d_ws;
  size_t off = 0;
  auto alloc = [&](size_t bytes) -> void* {
    void* p = wsb + off;
    off += (bytes + 255) & ~(size_t)255;
    return p;
  };
  float*          xssm_f  = (float*)         alloc((size_t)MTOK * Dm * 4);
  unsigned short* xssm_h  = (unsigned short*)alloc((size_t)MTOK * Dm * 2);
  float*          siluz   = (float*)         alloc((size_t)MTOK * Dm * 4);
  float*          xdbl    = (float*)         alloc((size_t)MTOK * XP * 4);
  unsigned short* dtlow_h = (unsigned short*)alloc((size_t)MTOK * Rr * 2);
  float*          dtb     = (float*)         alloc((size_t)MTOK * Dm * 4);
  float*          Yb      = (float*)         alloc((size_t)Bz * NCH * Dm * 4);
  float*          Wb      = (float*)         alloc((size_t)Bz * NCH * Ss * Dm * 4);
  float*          Pb      = (float*)         alloc((size_t)Bz * NCH * Ss * Dm * 4);
  float*          Hb      = (float*)         alloc((size_t)Bz * NCH * Ss * Dm * 4);
  float*          ctx     = (float*)         alloc((size_t)Bz * Dm * 4);
  float*          normed  = (float*)         alloc((size_t)Bz * Dm * 4);
  float*          outrow  = (float*)         alloc((size_t)Bz * Dm * 4);

  gemm1_inproj<<<dim3(2 * Dm / 128, MTOK / 128), 256, 0, stream>>>(
      values, in_proj_w, in_proj_b, xssm_f, xssm_h, siluz);
  gemm2_xproj<<<dim3(1, MTOK / 128), 256, 0, stream>>>(
      xssm_h, x_proj_w, x_proj_b, xdbl, dtlow_h);
  gemm3_dtproj<<<dim3(Dm / 128, MTOK / 128), 256, 0, stream>>>(
      dtlow_h, dt_proj_w, dt_proj_b, dtb);
  scan_part1<<<dim3(Dm / 256, NCH, Bz), 256, 0, stream>>>(
      dtb, xssm_f, siluz, xdbl, A_log, D_param, Yb, Wb, Pb, Hb);
  scan_part2<<<dim3(Bz * Dm / 256), 256, 0, stream>>>(Yb, Wb, Pb, Hb, ctx);
  ln_kernel<<<dim3(Bz), 256, 0, stream>>>(ctx, ln_gamma, ln_beta, normed);
  outproj_kernel<<<dim3(Bz * 4), 256, 0, stream>>>(normed, out_proj_w, out_proj_b, outrow);
  bcast_kernel<<<dim3(8192), 256, 0, stream>>>((const float4*)outrow, (float4*)out);
}
